// ModulatedDecoder_32573031973088
// MI455X (gfx1250) — compile-verified, matrix operand order checked by reading, not run on hardware
//
#include <hip/hip_runtime.h>

typedef __bf16 bf16_t;
typedef __attribute__((ext_vector_type(16))) __bf16 v16bf;
typedef __attribute__((ext_vector_type(8)))  float  v8f;
typedef __attribute__((ext_vector_type(4)))  unsigned int v4u;
typedef __attribute__((ext_vector_type(8)))  int  v8i;
typedef __attribute__((ext_vector_type(4)))  int  v4i;
typedef int gv4i __attribute__((vector_size(16)));   // matches builtin pointer pointee type

#define AS1 __attribute__((address_space(1)))
#define AS3 __attribute__((address_space(3)))

#define B_   32
#define T_   2000
#define H_   512
#define G3_  1536
#define NH_  100
#define NHP_ 112
#define NM_  65
#define NMP_ 80
#define E_   16
#define HH_  256
#define BT_  (B_*T_)

// ---- CDNA5 feature detection (guarded so the build never regresses) ----
#if defined(__has_builtin)
#  if __has_builtin(__builtin_amdgcn_tensor_load_to_lds) && __has_builtin(__builtin_amdgcn_s_wait_tensorcnt)
#    define USE_TDM 1
#  endif
#  if __has_builtin(__builtin_amdgcn_global_load_async_to_lds_b128) && __has_builtin(__builtin_amdgcn_s_wait_asynccnt)
#    define USE_ASYNC 1
#  endif
#endif
#ifndef USE_TDM
#define USE_TDM 0
#endif
#ifndef USE_ASYNC
#define USE_ASYNC 0
#endif

// ---------------- workspace layout ----------------
constexpr size_t ALN(size_t x){ return (x + 255) & ~size_t(255); }
constexpr size_t OFF_HBF   = 0;                                              // h bf16 [32,256]
constexpr size_t OFF_EHB   = ALN(OFF_HBF   + (size_t)B_*HH_*2);              // eff harm bias [32,112] f32
constexpr size_t OFF_ENB   = ALN(OFF_EHB   + (size_t)B_*NHP_*4);             // eff noise bias [32,80] f32
constexpr size_t OFF_EHW   = ALN(OFF_ENB   + (size_t)B_*NMP_*4);             // eff harm W bf16 [32,112,512]
constexpr size_t OFF_ENW   = ALN(OFF_EHW   + (size_t)B_*NHP_*H_*2);          // eff noise W bf16 [32,80,512]
constexpr size_t OFF_WIHB  = ALN(OFF_ENW   + (size_t)B_*NMP_*H_*2);          // W_ih bf16
constexpr size_t OFF_WHHB  = ALN(OFF_WIHB  + (size_t)G3_*H_*2);              // W_hh bf16
constexpr size_t OFF_WPB   = ALN(OFF_WHHB  + (size_t)G3_*H_*2);              // post_W bf16
constexpr size_t OFF_XPRE  = ALN(OFF_WPB   + (size_t)H_*H_*2);               // xpre bf16 [BT,512]
constexpr size_t OFF_GI    = ALN(OFF_XPRE  + (size_t)BT_*H_*2);              // gi bf16 [BT,1536]
constexpr size_t OFF_Y     = ALN(OFF_GI    + (size_t)BT_*G3_*2);             // gru out bf16 [BT,512]
constexpr size_t OFF_HST32 = ALN(OFF_Y     + (size_t)BT_*H_*2);              // h state f32 x2 [32,512]
constexpr size_t OFF_HSTBF = ALN(OFF_HST32 + (size_t)2*B_*H_*4);             // h state bf16 x2
constexpr size_t OFF_BAR   = ALN(OFF_HSTBF + (size_t)2*B_*H_*2);             // grid barrier

#define GRU_NWG   8
#define GRU_LDSB  (3*64*512*2 + 32*512*2)    // 229376 B: W_hh slice + h stage (needs CDNA5 320KB WGP LDS)
#define GI_LDSB   (80*512*2)                 // 81920 B: A tile
#define POST_LDSB (2*80*512*2)               // 163840 B: A tile + y2 tile

// ---------------- helpers ----------------
__device__ inline float sigmoidf_(float x){ return 1.f/(1.f + __expf(-x)); }
__device__ inline float leakyf_(float x){ return x > 0.f ? x : 0.01f*x; }
__device__ inline bf16_t tobf(float f){ return (bf16_t)f; }

union FragU { uint4 u[2]; v16bf f; };

// bf16 A/B fragment: row-major rowbase, 16 contiguous-K halves split per ISA layout.
__device__ inline v16bf load_frag_bf16(const bf16_t* rowbase, int k0, int half){
  FragU t;
  t.u[0] = *reinterpret_cast<const uint4*>(rowbase + k0 + half*8);
  t.u[1] = *reinterpret_cast<const uint4*>(rowbase + k0 + 16 + half*8);
  return t.f;
}
// same but source is f32 (convert on the fly)
__device__ inline v16bf load_frag_f32(const float* rowbase, int k0, int half){
  v16bf r;
  const float* p0 = rowbase + k0 + half*8;
  const float* p1 = rowbase + k0 + 16 + half*8;
#pragma unroll
  for (int i = 0; i < 8; ++i) r[i] = (__bf16)p0[i];
#pragma unroll
  for (int i = 0; i < 8; ++i) r[8+i] = (__bf16)p1[i];
  return r;
}
__device__ inline v8f wmma_bf(v16bf a, v16bf b, v8f c){
  return __builtin_amdgcn_wmma_f32_16x16x32_bf16(false, a, false, b, (short)0, c, false, false);
}
__device__ inline v8f zero8(){ v8f z = {0.f,0.f,0.f,0.f,0.f,0.f,0.f,0.f}; return z; }

__device__ inline unsigned lds_byte_addr(const void* p){
  return (unsigned)(size_t)(const AS3 char*)p;
}

#if USE_TDM
// One-wave TDM copy of a contiguous block (n8 units of 8 bytes, n8 <= 65535) into LDS.
// D# built per CDNA5 ISA: 2D tile with a single row.
__device__ inline void tdm_copy_1d(unsigned lds_addr, const void* gsrc, unsigned n8){
  unsigned long long ga = (unsigned long long)(size_t)gsrc;
  v4u g0;
  g0[0] = 1u;                                               // count=1, user mode, no gather
  g0[1] = lds_addr;                                         // lds_addr [63:32]
  g0[2] = (unsigned)(ga & 0xFFFFFFFFu);                     // global_addr [95:64]
  g0[3] = (unsigned)((ga >> 32) & 0x01FFFFFFu) | (2u << 30);// addr [120:96] | type=2
  v8i g1;
  g1[0] = (int)(3u << 16);                                  // data_size=3 (8B units)
  g1[1] = (int)((n8 & 0xFFFFu) << 16);                      // tensor_dim0[15:0]
  g1[2] = (int)(((n8 >> 16) & 0xFFFFu) | (1u << 16));       // tensor_dim0[31:16] | tensor_dim1=1
  g1[3] = (int)((n8 & 0xFFFFu) << 16);                      // tile_dim0
  g1[4] = 1;                                                // tile_dim1=1, tile_dim2=0
  g1[5] = (int)n8;                                          // tensor_dim0_stride[31:0]
  g1[6] = 0;
  g1[7] = 0;
  v4i gz; gz[0]=0; gz[1]=0; gz[2]=0; gz[3]=0;               // groups 2/3 unused (<=2D)
  v8i gz8; gz8[0]=0; gz8[1]=0; gz8[2]=0; gz8[3]=0; gz8[4]=0; gz8[5]=0; gz8[6]=0; gz8[7]=0;
  __builtin_amdgcn_tensor_load_to_lds(g0, g1, gz, gz, gz8, 0);
}
#endif

// Block-wide stage of a contiguous bf16 region into LDS (TDM > async-LDS > VGPR copy).
__device__ inline void stage_block(bf16_t* lds_dst, const bf16_t* gsrc, int n_elems){
#if USE_TDM
  if (threadIdx.x < 32){
    tdm_copy_1d(lds_byte_addr(lds_dst), (const void*)gsrc, (unsigned)(n_elems*2/8));
    __builtin_amdgcn_s_wait_tensorcnt(0);
  }
#elif USE_ASYNC
  for (int c = threadIdx.x; c < n_elems/8; c += blockDim.x){
    __builtin_amdgcn_global_load_async_to_lds_b128(
        (AS1 gv4i*)(gsrc + (size_t)c*8), (AS3 gv4i*)(lds_dst + (size_t)c*8), 0, 0);
  }
  __builtin_amdgcn_s_wait_asynccnt(0);
#else
  {
    const uint4* src = reinterpret_cast<const uint4*>(gsrc);
    uint4* dst = reinterpret_cast<uint4*>(lds_dst);
    for (int c = threadIdx.x; c < n_elems/8; c += blockDim.x) dst[c] = src[c];
  }
#endif
}

// device-scope grid barrier (persistent-kernel sync, one per GRU step)
__device__ inline void grid_barrier(unsigned* bar, unsigned nwg){
  __builtin_amdgcn_fence(__ATOMIC_RELEASE, "agent");
  __syncthreads();
  if (threadIdx.x == 0){
    unsigned gen  = __atomic_load_n(bar + 1, __ATOMIC_RELAXED);
    unsigned prev = __atomic_fetch_add(bar, 1u, __ATOMIC_ACQ_REL);
    if (prev == nwg - 1u){
      __atomic_store_n(bar, 0u, __ATOMIC_RELAXED);
      __atomic_store_n(bar + 1, gen + 1u, __ATOMIC_RELEASE);
    } else {
      while (__atomic_load_n(bar + 1, __ATOMIC_ACQUIRE) == gen){
        __builtin_amdgcn_s_sleep(2);
      }
    }
  }
  __syncthreads();
  __builtin_amdgcn_fence(__ATOMIC_ACQUIRE, "agent");
}

// ---------------- kernel 0: init state / barrier / pad rows ----------------
__global__ void k_init(float* hst32, bf16_t* hstbf, bf16_t* ehw, bf16_t* enw, unsigned* bar){
  int tid = blockIdx.x*blockDim.x + threadIdx.x;
  int stride = gridDim.x*blockDim.x;
  if (tid < 64) bar[tid] = 0u;
  for (int i = tid; i < 2*B_*H_; i += stride) hst32[i] = 0.f;
  for (int i = tid; i < 2*B_*H_; i += stride) hstbf[i] = tobf(0.f);
  for (int i = tid; i < B_*12*H_; i += stride){            // eff harm pad rows o in [100,112)
    int b = i/(12*H_); int r = i%(12*H_);
    ehw[((size_t)b*NHP_ + NH_)*H_ + r] = tobf(0.f);
  }
  for (int i = tid; i < B_*15*H_; i += stride){            // eff noise pad rows o in [65,80)
    int b = i/(15*H_); int r = i%(15*H_);
    enw[((size_t)b*NMP_ + NM_)*H_ + r] = tobf(0.f);
  }
}

// ---------------- kernel: f32 -> bf16 weight conversion ----------------
__global__ void k_convert(const float* __restrict__ src, bf16_t* __restrict__ dst, int n){
  int tid = blockIdx.x*blockDim.x + threadIdx.x;
  int stride = gridDim.x*blockDim.x;
  for (int i = tid; i < n; i += stride) dst[i] = tobf(src[i]);
}

// ---------------- kernel 1: hypernetwork h + effective head biases ----------------
__global__ void k_hyper(const float* __restrict__ es,
                        const float* __restrict__ W1, const float* __restrict__ b1,
                        const float* __restrict__ Wbh, const float* __restrict__ bbh,
                        const float* __restrict__ Wbn, const float* __restrict__ bbn,
                        const float* __restrict__ harm_b, const float* __restrict__ noise_b,
                        bf16_t* __restrict__ hbf, float* __restrict__ ehb, float* __restrict__ enb){
  __shared__ float es_s[B_*E_];
  __shared__ float h_s[B_*HH_];
  int tid = threadIdx.x;
  if (tid < B_*E_) es_s[tid] = es[tid];
  __syncthreads();
  int j = tid;                       // 256 threads == HH_
  for (int b = 0; b < B_; ++b){
    float s = b1[j];
#pragma unroll
    for (int k = 0; k < E_; ++k) s += es_s[b*E_ + k] * W1[j*E_ + k];
    s = s > 0.f ? s : 0.f;           // relu
    h_s[b*HH_ + j] = s;
    hbf[b*HH_ + j] = tobf(s);
  }
  __syncthreads();
  for (int idx = tid; idx < B_*NHP_ + B_*NMP_; idx += blockDim.x){
    if (idx < B_*NHP_){
      int b = idx / NHP_, o = idx % NHP_;
      float v = 0.f;
      if (o < NH_){
        v = harm_b[o] + bbh[o];
        for (int k = 0; k < HH_; ++k) v += h_s[b*HH_ + k] * Wbh[o*HH_ + k];
      }
      ehb[idx] = v;
    } else {
      int i2 = idx - B_*NHP_;
      int b = i2 / NMP_, o = i2 % NMP_;
      float v = 0.f;
      if (o < NM_){
        v = noise_b[o] + bbn[o];
        for (int k = 0; k < HH_; ++k) v += h_s[b*HH_ + k] * Wbn[o*HH_ + k];
      }
      enb[i2] = v;
    }
  }
}

// ---------------- kernel 2: effective head weights (WMMA, [32,256] x [256,N]) ----------------
__global__ void k_effw(const bf16_t* __restrict__ hbf,
                       const float* __restrict__ Wdh, const float* __restrict__ bdh,
                       const float* __restrict__ Wdn, const float* __restrict__ bdn,
                       const float* __restrict__ harm_W, const float* __restrict__ noise_W,
                       bf16_t* __restrict__ ehw, bf16_t* __restrict__ enw){
  int lane = threadIdx.x & 31, half = lane >> 4, ln = lane & 15;
  int gw = blockIdx.x*8 + (threadIdx.x >> 5);
  bool harm = gw < (NH_*H_)/16;
  int nt = harm ? gw : gw - (NH_*H_)/16;
  int c0 = nt*16;
  const float* Wd = harm ? Wdh : Wdn;
  const bf16_t* pA0 = hbf + (size_t)ln*HH_;
  const bf16_t* pA1 = hbf + (size_t)(16 + ln)*HH_;
  const float*  pB  = Wd + (size_t)(c0 + ln)*HH_;
  v8f acc0 = zero8(), acc1 = zero8();
#pragma unroll
  for (int kk = 0; kk < HH_/32; ++kk){
    int k0 = kk*32;
    v16bf bfr = load_frag_f32(pB, k0, half);
    acc0 = wmma_bf(load_frag_bf16(pA0, k0, half), bfr, acc0);
    acc1 = wmma_bf(load_frag_bf16(pA1, k0, half), bfr, acc1);
  }
  int c = c0 + ln;
  if (harm){
    int o = c / H_, k = c % H_;
    float base = harm_W[c] + bdh[c];
#pragma unroll
    for (int r = 0; r < 8; ++r){
      int b0 = r + 8*half;
      ehw[((size_t)b0*NHP_ + o)*H_ + k]        = tobf(base + acc0[r]);
      ehw[((size_t)(16 + b0)*NHP_ + o)*H_ + k] = tobf(base + acc1[r]);
    }
  } else {
    int o = c / H_, k = c % H_;
    float base = noise_W[c] + bdn[c];
#pragma unroll
    for (int r = 0; r < 8; ++r){
      int b0 = r + 8*half;
      enw[((size_t)b0*NMP_ + o)*H_ + k]        = tobf(base + acc0[r]);
      enw[((size_t)(16 + b0)*NMP_ + o)*H_ + k] = tobf(base + acc1[r]);
    }
  }
}

// ---------------- kernel 3a: pre-layer (K=2, elementwise) ----------------
__global__ void k_xpre(const float* __restrict__ f0, const float* __restrict__ loud,
                       const float* __restrict__ pre_W, const float* __restrict__ pre_b,
                       bf16_t* __restrict__ xpre){
  size_t tid = (size_t)blockIdx.x*blockDim.x + threadIdx.x;
  size_t stride = (size_t)gridDim.x*blockDim.x;
  for (size_t i = tid; i < (size_t)BT_*H_; i += stride){
    size_t m = i / H_; int j = (int)(i % H_);
    float v = f0[m]*pre_W[j*2+0] + loud[m]*pre_W[j*2+1] + pre_b[j];
    xpre[i] = tobf(leakyf_(v));
  }
}

// ---------------- kernel 3b: gi = xpre @ W_ih^T + b_ih  (WMMA, M=64000 N=1536 K=512) ----------------
__global__ void k_gi(const bf16_t* __restrict__ xpre, const bf16_t* __restrict__ wihb,
                     const float* __restrict__ b_ih, bf16_t* __restrict__ gi){
  extern __shared__ bf16_t sm_gi[];           // A tile [80][512]
  int r0 = blockIdx.x * 80;
  stage_block(sm_gi, xpre + (size_t)r0*H_, 80*H_);   // TDM / async / copy
  __syncthreads();
  int lane = threadIdx.x & 31, half = lane >> 4, ln = lane & 15;
  int wv = threadIdx.x >> 5;
  for (int g = wv; g < 5*24; g += 8){         // 5 M-tiles x 24 groups of 4 N-tiles
    int mt = g / 24, ng = g % 24, nt0 = ng*4;
    const bf16_t* pA = sm_gi + (size_t)(mt*16 + ln)*H_;
    const bf16_t* pB0 = wihb + (size_t)((nt0+0)*16 + ln)*H_;
    const bf16_t* pB1 = wihb + (size_t)((nt0+1)*16 + ln)*H_;
    const bf16_t* pB2 = wihb + (size_t)((nt0+2)*16 + ln)*H_;
    const bf16_t* pB3 = wihb + (size_t)((nt0+3)*16 + ln)*H_;
    v8f a0 = zero8(), a1 = zero8(), a2 = zero8(), a3 = zero8();
#pragma unroll 4
    for (int kk = 0; kk < H_/32; ++kk){
      int k0 = kk*32;
      __builtin_prefetch((const void*)(pB0 + k0 + 64), 0, 1);
      v16bf af = load_frag_bf16(pA, k0, half);
      a0 = wmma_bf(af, load_frag_bf16(pB0, k0, half), a0);
      a1 = wmma_bf(af, load_frag_bf16(pB1, k0, half), a1);
      a2 = wmma_bf(af, load_frag_bf16(pB2, k0, half), a2);
      a3 = wmma_bf(af, load_frag_bf16(pB3, k0, half), a3);
    }
#pragma unroll
    for (int i = 0; i < 4; ++i){
      v8f acc = (i==0)?a0:(i==1)?a1:(i==2)?a2:a3;
      int col = (nt0+i)*16 + ln;
      float bias = b_ih[col];
#pragma unroll
      for (int r = 0; r < 8; ++r){
        int m = mt*16 + r + 8*half;
        gi[(size_t)(r0 + m)*G3_ + col] = tobf(acc[r] + bias);
      }
    }
  }
}

// ---------------- kernel 4: persistent GRU (8 WGPs, W_hh slice resident in LDS) ----------------
__global__ void k_gru(const bf16_t* __restrict__ whhb, const bf16_t* __restrict__ gi,
                      const float* __restrict__ b_hh,
                      float* __restrict__ hst32, bf16_t* __restrict__ hstbf,
                      bf16_t* __restrict__ ybf, unsigned* bar){
  extern __shared__ char sm_gru[];
  bf16_t* Wl = reinterpret_cast<bf16_t*>(sm_gru);               // [3*64][512]
  bf16_t* hl = reinterpret_cast<bf16_t*>(sm_gru + 3*64*H_*2);   // [32][512]
  int j0w = blockIdx.x * 64;

  // stage W_hh slice once: rows {g*512 + j0w + u : g<3, u<64}
#if USE_ASYNC
  for (int c = threadIdx.x; c < (3*64*H_)/8; c += blockDim.x){
    int rowL = c >> 6;
    int kc   = (c & 63) * 8;
    int g = rowL / 64, u = rowL % 64;
    __builtin_amdgcn_global_load_async_to_lds_b128(
        (AS1 gv4i*)(whhb + ((size_t)(g*H_ + j0w + u))*H_ + kc),
        (AS3 gv4i*)(Wl + (size_t)rowL*H_ + kc), 0, 0);
  }
  __builtin_amdgcn_s_wait_asynccnt(0);
#else
  for (int c = threadIdx.x; c < (3*64*H_)/8; c += blockDim.x){
    int rowL = c >> 6;                // 64 uint4 chunks per 512-elem row
    int kc   = (c & 63) * 8;
    int g = rowL / 64, u = rowL % 64;
    reinterpret_cast<uint4*>(Wl)[c] =
      *reinterpret_cast<const uint4*>(whhb + ((size_t)(g*H_ + j0w + u))*H_ + kc);
  }
#endif
  __syncthreads();

  int lane = threadIdx.x & 31, half = lane >> 4, ln = lane & 15;
  int wv = threadIdx.x >> 5, mt = wv >> 2, jt = wv & 3;
  int j = j0w + jt*16 + ln;                       // hidden unit for this lane's C column
  float bhr = b_hh[j], bhz = b_hh[H_ + j], bhn = b_hh[2*H_ + j];
  const bf16_t* pA  = hl + (size_t)(mt*16 + ln)*H_;
  const bf16_t* pBr = Wl + (size_t)(0*64 + jt*16 + ln)*H_;
  const bf16_t* pBz = Wl + (size_t)(1*64 + jt*16 + ln)*H_;
  const bf16_t* pBn = Wl + (size_t)(2*64 + jt*16 + ln)*H_;

  for (int t = 0; t < T_; ++t){
    int cur = t & 1, nxt = cur ^ 1;
    // stage h_t (bf16) into LDS: TDM DMA when available
    stage_block(hl, hstbf + (size_t)cur*B_*H_, B_*H_);
    __syncthreads();
    v8f aR = zero8(), aZ = zero8(), aN = zero8();
#pragma unroll 4
    for (int kk = 0; kk < H_/32; ++kk){
      int k0 = kk*32;
      v16bf af = load_frag_bf16(pA, k0, half);
      aR = wmma_bf(af, load_frag_bf16(pBr, k0, half), aR);
      aZ = wmma_bf(af, load_frag_bf16(pBz, k0, half), aZ);
      aN = wmma_bf(af, load_frag_bf16(pBn, k0, half), aN);
    }
#pragma unroll
    for (int r = 0; r < 8; ++r){
      int m = mt*16 + r + 8*half;                 // batch index
      size_t gr = ((size_t)m*T_ + t)*G3_;
      float gir = (float)gi[gr + j];
      float giz = (float)gi[gr + H_ + j];
      float gin = (float)gi[gr + 2*H_ + j];
      float rg = sigmoidf_(gir + aR[r] + bhr);
      float zg = sigmoidf_(giz + aZ[r] + bhz);
      float ng = tanhf(gin + rg*(aN[r] + bhn));
      float hp = hst32[(size_t)cur*B_*H_ + (size_t)m*H_ + j];
      float hn = (1.f - zg)*ng + zg*hp;
      hst32[(size_t)nxt*B_*H_ + (size_t)m*H_ + j] = hn;
      hstbf[(size_t)nxt*B_*H_ + (size_t)m*H_ + j] = tobf(hn);
      ybf[((size_t)m*T_ + t)*H_ + j] = tobf(hn);
    }
    grid_barrier(bar, GRU_NWG);
  }
}

// ---------------- kernel 5: post layer + modulated heads (fused, WMMA) ----------------
__global__ void k_post(const bf16_t* __restrict__ ybf, const bf16_t* __restrict__ wpb,
                       const float* __restrict__ post_b,
                       const bf16_t* __restrict__ ehw, const bf16_t* __restrict__ enw,
                       const float* __restrict__ ehb, const float* __restrict__ enb,
                       float* __restrict__ out){
  extern __shared__ bf16_t sm_post[];
  bf16_t* As = sm_post;                 // y tile  [80][512]
  bf16_t* Y2 = sm_post + 80*H_;         // y2 tile [80][512]
  int b = blockIdx.x / 25;
  int t0 = (blockIdx.x % 25) * 80;
  stage_block(As, ybf + ((size_t)b*T_ + t0)*H_, 80*H_);   // TDM / async / copy
  __syncthreads();
  int lane = threadIdx.x & 31, half = lane >> 4, ln = lane & 15;
  int wv = threadIdx.x >> 5;
  // phase 1: y2 = leaky(y @ post_W^T + post_b), kept in LDS as bf16
  for (int g = wv; g < 5*8; g += 8){    // 5 M-tiles x 8 groups of 4 N-tiles
    int mt = g / 8, nt0 = (g % 8)*4;
    const bf16_t* pA = As + (size_t)(mt*16 + ln)*H_;
    const bf16_t* pB0 = wpb + (size_t)((nt0+0)*16 + ln)*H_;
    const bf16_t* pB1 = wpb + (size_t)((nt0+1)*16 + ln)*H_;
    const bf16_t* pB2 = wpb + (size_t)((nt0+2)*16 + ln)*H_;
    const bf16_t* pB3 = wpb + (size_t)((nt0+3)*16 + ln)*H_;
    v8f a0 = zero8(), a1 = zero8(), a2 = zero8(), a3 = zero8();
#pragma unroll 4
    for (int kk = 0; kk < H_/32; ++kk){
      int k0 = kk*32;
      __builtin_prefetch((const void*)(pB0 + k0 + 64), 0, 1);
      v16bf af = load_frag_bf16(pA, k0, half);
      a0 = wmma_bf(af, load_frag_bf16(pB0, k0, half), a0);
      a1 = wmma_bf(af, load_frag_bf16(pB1, k0, half), a1);
      a2 = wmma_bf(af, load_frag_bf16(pB2, k0, half), a2);
      a3 = wmma_bf(af, load_frag_bf16(pB3, k0, half), a3);
    }
#pragma unroll
    for (int i = 0; i < 4; ++i){
      v8f acc = (i==0)?a0:(i==1)?a1:(i==2)?a2:a3;
      int col = (nt0+i)*16 + ln;
      float bias = post_b[col];
#pragma unroll
      for (int r = 0; r < 8; ++r){
        int m = mt*16 + r + 8*half;
        Y2[(size_t)m*H_ + col] = tobf(leakyf_(acc[r] + bias));
      }
    }
  }
  __syncthreads();
  // phase 2: heads with per-batch effective weights (35 harm tiles + 25 noise tiles)
  float* out_h = out;
  float* out_n = out + (size_t)B_*T_*NH_;
  for (int idx = wv; idx < 35 + 25; idx += 8){
    bool harm = idx < 35;
    int mt, nt;
    const bf16_t* effW;
    if (harm){ mt = idx / 7; nt = idx % 7; effW = ehw + (size_t)b*NHP_*H_; }
    else     { int i2 = idx - 35; mt = i2 / 5; nt = i2 % 5; effW = enw + (size_t)b*NMP_*H_; }
    const bf16_t* pA = Y2 + (size_t)(mt*16 + ln)*H_;
    const bf16_t* pB = effW + (size_t)(nt*16 + ln)*H_;
    v8f acc = zero8();
#pragma unroll 4
    for (int kk = 0; kk < H_/32; ++kk){
      int k0 = kk*32;
      acc = wmma_bf(load_frag_bf16(pA, k0, half), load_frag_bf16(pB, k0, half), acc);
    }
    int o = nt*16 + ln;
    if (harm){
      if (o < NH_){
        float bias = ehb[b*NHP_ + o];
#pragma unroll
        for (int r = 0; r < 8; ++r){
          int tt = t0 + mt*16 + r + 8*half;
          out_h[((size_t)b*T_ + tt)*NH_ + o] = sigmoidf_(acc[r] + bias);
        }
      }
    } else {
      if (o < NM_){
        float bias = enb[b*NMP_ + o];
#pragma unroll
        for (int r = 0; r < 8; ++r){
          int tt = t0 + mt*16 + r + 8*half;
          out_n[((size_t)b*T_ + tt)*NM_ + o] = sigmoidf_(acc[r] + bias);
        }
      }
    }
  }
}

// ---------------- host launcher ----------------
extern "C" void kernel_launch(void* const* d_in, const int* in_sizes, int n_in,
                              void* d_out, int out_size, void* d_ws, size_t ws_size,
                              hipStream_t stream){
  (void)in_sizes; (void)n_in; (void)out_size; (void)ws_size;
  const float* f0      = (const float*)d_in[0];
  const float* loud    = (const float*)d_in[1];
  const float* es      = (const float*)d_in[2];
  const float* pre_W   = (const float*)d_in[3];
  const float* pre_b   = (const float*)d_in[4];
  const float* W_ih    = (const float*)d_in[5];
  const float* W_hh    = (const float*)d_in[6];
  const float* b_ih    = (const float*)d_in[7];
  const float* b_hh    = (const float*)d_in[8];
  const float* post_W  = (const float*)d_in[9];
  const float* post_b  = (const float*)d_in[10];
  const float* harm_W  = (const float*)d_in[11];
  const float* harm_b  = (const float*)d_in[12];
  const float* noise_W = (const float*)d_in[13];
  const float* noise_b = (const float*)d_in[14];
  const float* hyp_W1  = (const float*)d_in[15];
  const float* hyp_b1  = (const float*)d_in[16];
  const float* hyp_Wdh = (const float*)d_in[17];
  const float* hyp_bdh = (const float*)d_in[18];
  const float* hyp_Wbh = (const float*)d_in[19];
  const float* hyp_bbh = (const float*)d_in[20];
  const float* hyp_Wdn = (const float*)d_in[21];
  const float* hyp_bdn = (const float*)d_in[22];
  const float* hyp_Wbn = (const float*)d_in[23];
  const float* hyp_bbn = (const float*)d_in[24];

  char* ws = (char*)d_ws;
  bf16_t*  hbf   = (bf16_t*)(ws + OFF_HBF);
  float*   ehb   = (float*) (ws + OFF_EHB);
  float*   enb   = (float*) (ws + OFF_ENB);
  bf16_t*  ehw   = (bf16_t*)(ws + OFF_EHW);
  bf16_t*  enw   = (bf16_t*)(ws + OFF_ENW);
  bf16_t*  wihb  = (bf16_t*)(ws + OFF_WIHB);
  bf16_t*  whhb  = (bf16_t*)(ws + OFF_WHHB);
  bf16_t*  wpb   = (bf16_t*)(ws + OFF_WPB);
  bf16_t*  xpre  = (bf16_t*)(ws + OFF_XPRE);
  bf16_t*  gi    = (bf16_t*)(ws + OFF_GI);
  bf16_t*  ybf   = (bf16_t*)(ws + OFF_Y);
  float*   hst32 = (float*) (ws + OFF_HST32);
  bf16_t*  hstbf = (bf16_t*)(ws + OFF_HSTBF);
  unsigned* bar  = (unsigned*)(ws + OFF_BAR);
  float*   out   = (float*)d_out;

  (void)hipFuncSetAttribute((const void*)k_gi,   hipFuncAttributeMaxDynamicSharedMemorySize, GI_LDSB);
  (void)hipFuncSetAttribute((const void*)k_gru,  hipFuncAttributeMaxDynamicSharedMemorySize, GRU_LDSB);
  (void)hipFuncSetAttribute((const void*)k_post, hipFuncAttributeMaxDynamicSharedMemorySize, POST_LDSB);

  k_init<<<64, 256, 0, stream>>>(hst32, hstbf, ehw, enw, bar);
  k_convert<<<512, 256, 0, stream>>>(W_ih,   wihb, G3_*H_);
  k_convert<<<512, 256, 0, stream>>>(W_hh,   whhb, G3_*H_);
  k_convert<<<256, 256, 0, stream>>>(post_W, wpb,  H_*H_);
  k_hyper<<<1, 256, 0, stream>>>(es, hyp_W1, hyp_b1, hyp_Wbh, hyp_bbh, hyp_Wbn, hyp_bbn,
                                 harm_b, noise_b, hbf, ehb, enb);
  k_effw<<<660, 256, 0, stream>>>(hbf, hyp_Wdh, hyp_bdh, hyp_Wdn, hyp_bdn,
                                  harm_W, noise_W, ehw, enw);
  k_xpre<<<4096, 256, 0, stream>>>(f0, loud, pre_W, pre_b, xpre);
  k_gi<<<800, 256, GI_LDSB, stream>>>(xpre, wihb, b_ih, gi);
  k_gru<<<GRU_NWG, 256, GRU_LDSB, stream>>>(whhb, gi, b_hh, hst32, hstbf, ybf, bar);
  k_post<<<800, 256, POST_LDSB, stream>>>(ybf, wpb, post_b, ehw, enw, ehb, enb, out);
}
